// rpn_23622320128429
// MI455X (gfx1250) — matrix operand order repaired and checked by reading, not compiled, not run on hardware
//
#include <hip/hip_runtime.h>
#include <hip/hip_bf16.h>

typedef __attribute__((ext_vector_type(16))) __bf16 bf16x16;
typedef __attribute__((ext_vector_type(8)))  float  f32x8;

#define HW      10000
#define CIN     512
#define KDIM    4608
#define NANCH   90000
#define NPAD    131072
#define PRENMS  6000
#define NWRD    188          // ceil(6000/32)
#define IMGSZ   1600.0f

__constant__ float c_sizes[9] = {4.f, 8.f, 16.f, 24.f, 32.f, 48.f, 64.f, 80.f, 96.f};

union Frag { bf16x16 v; uint4 q[2]; };

__device__ __forceinline__ unsigned short f2bf(float f) {
  unsigned int u = __float_as_uint(f);
  unsigned int r = u + 0x7FFFu + ((u >> 16) & 1u);
  return (unsigned short)(r >> 16);
}

__device__ __forceinline__ unsigned fmono(float f) {
  unsigned u = __float_as_uint(f);
  return (u & 0x80000000u) ? ~u : (u | 0x80000000u);
}

__device__ __forceinline__ void anchor_box(int n, float& x1, float& y1, float& x2, float& y2) {
  int a = n % 9, m = n / 9;
  float s = c_sizes[a];
  float cx = (float)(m / 100) * 16.f + 8.f;
  float cy = (float)(m % 100) * 16.f + 8.f;
  x1 = cx - s * 0.5f; y1 = cy - s * 0.5f;
  x2 = cx + s * 0.5f; y2 = cy + s * 0.5f;
}

// async global->LDS copy of 16B per lane (gfx1250 TDM-adjacent path, ASYNCcnt tracked)
__device__ __forceinline__ void async_ld16(unsigned lds_addr, const void* gaddr) {
  asm volatile("global_load_async_to_lds_b128 %0, %1, off"
               :: "v"(lds_addr), "v"(gaddr) : "memory");
}

// ---------------- prep kernels ----------------

__global__ void zero_misc_kernel(unsigned long long* gkey, float* accum) {
  int t = threadIdx.x;
  if (t < 64) gkey[t] = 0ull;
  if (t < 4)  accum[t] = 0.f;
}

// feat (1,512,100,100) f32 -> padded HWC bf16 [102][102][512]
__global__ void pad_feat_kernel(const float* __restrict__ feat, unsigned short* __restrict__ fp) {
  int idx = blockIdx.x * 256 + threadIdx.x;
  if (idx >= 102 * 102 * 512) return;
  int c = idx & 511, p = idx >> 9;
  int y = p / 102, x = p % 102;
  float v = 0.f;
  if (y >= 1 && y <= 100 && x >= 1 && x <= 100)
    v = feat[c * 10000 + (y - 1) * 100 + (x - 1)];
  fp[idx] = f2bf(v);
}

// conv_w (512,512,3,3) -> Bt bf16 [oc][k], k = (kh*3+kw)*512 + ic
__global__ void prep_bt_kernel(const float* __restrict__ w, unsigned short* __restrict__ Bt) {
  int idx = blockIdx.x * 256 + threadIdx.x;
  if (idx >= 512 * KDIM) return;
  int oc = idx / KDIM, k = idx % KDIM;
  int khw = k >> 9, ic = k & 511;
  int kh = khw / 3, kw = khw % 3;
  Bt[idx] = f2bf(w[((oc * 512 + ic) * 3 + kh) * 3 + kw]);
}

// pack score_w(18,512)+delta_w(36,512)+pad -> Wc bf16 [64][512]; bias cat -> bcat f32[64]
__global__ void prep_wcat_kernel(const float* __restrict__ sw, const float* __restrict__ sb,
                                 const float* __restrict__ dw, const float* __restrict__ db,
                                 unsigned short* __restrict__ Wc, float* __restrict__ bcat) {
  int idx = blockIdx.x * 256 + threadIdx.x;
  if (idx >= 64 * 512) return;
  int r = idx >> 9, ic = idx & 511;
  float v = 0.f;
  if (r < 18) v = sw[r * 512 + ic];
  else if (r < 54) v = dw[(r - 18) * 512 + ic];
  Wc[idx] = f2bf(v);
  if (idx < 64) {
    float b = 0.f;
    if (idx < 18) b = sb[idx];
    else if (idx < 54) b = db[idx - 18];
    bcat[idx] = b;
  }
}

// ---------------- WMMA GEMMs ----------------

// implicit-GEMM 3x3 conv: M=10000 (spatial), N=512 (oc), K=4608. bias+ReLU -> xact bf16
// Double-buffered LDS, GLOBAL_LOAD_ASYNC_TO_LDS_B128 pipelined 2 tiles ahead.
__launch_bounds__(256)
__global__ void conv_gemm_kernel(const unsigned short* __restrict__ fp,
                                 const unsigned short* __restrict__ Bt,
                                 const float* __restrict__ bias,
                                 unsigned short* __restrict__ xact) {
  __shared__ __align__(16) unsigned short As[2][64 * 64];
  __shared__ __align__(16) unsigned short Bs[2][64 * 64];
  const int t  = threadIdx.x;
  const int mb = blockIdx.x * 64;
  const int nb = blockIdx.y * 64;
  const int ra = t >> 2;
  const int c8 = (t & 3) * 8;          // 16B chunk (elements)
  int m = mb + ra; if (m >= HW) m = HW - 1;
  const int h = m / 100, w = m % 100;
  const int nrow = nb + ra;

  const int wave = t >> 5, lane = t & 31;
  const int wm = wave >> 2, wn = wave & 3;
  const int hb = lane >> 4, lr = lane & 15;

  const unsigned lA0 = (unsigned)(size_t)&As[0][0] + (unsigned)((ra * 64 + c8) * 2);
  const unsigned lB0 = (unsigned)(size_t)&Bs[0][0] + (unsigned)((ra * 64 + c8) * 2);

  auto issue = [&](int kk, int buf) {
    const int k0  = kk * 64;
    const int khw = k0 >> 9;
    const int kh  = khw / 3, kw = khw % 3;
    const int ic0 = k0 & 511;
    const unsigned short* ga = fp + ((h + kh) * 102 + (w + kw)) * 512 + ic0 + c8;
    const unsigned short* gb = Bt + nrow * KDIM + k0 + c8;
    const unsigned la = lA0 + (unsigned)buf * 8192u;
    const unsigned lb = lB0 + (unsigned)buf * 8192u;
    async_ld16(la,       ga);
    async_ld16(la + 64u, ga + 32);
    async_ld16(lb,       gb);
    async_ld16(lb + 64u, gb + 32);
  };

  f32x8 acc0 = {}; f32x8 acc1 = {};
  issue(0, 0);
  issue(1, 1);

  for (int kk = 0; kk < 72; ++kk) {
    if (kk < 71) { asm volatile("s_wait_asynccnt 4" ::: "memory"); }
    else         { asm volatile("s_wait_asynccnt 0" ::: "memory"); }
    __syncthreads();
    const unsigned short* Ab = &As[kk & 1][0];
    const unsigned short* Bb = &Bs[kk & 1][0];
#pragma unroll
    for (int ks = 0; ks < 64; ks += 32) {
      Frag a0, a1, b;
      const unsigned short* ar0 = Ab + (wm * 32 + lr) * 64 + ks;
      const unsigned short* ar1 = Ab + (wm * 32 + 16 + lr) * 64 + ks;
      const unsigned short* br  = Bb + (wn * 16 + lr) * 64 + ks;
      a0.q[0] = *(const uint4*)(ar0 + hb * 8);
      a0.q[1] = *(const uint4*)(ar0 + hb * 8 + 16);
      a1.q[0] = *(const uint4*)(ar1 + hb * 8);
      a1.q[1] = *(const uint4*)(ar1 + hb * 8 + 16);
      b.q[0]  = *(const uint4*)(br + hb * 8);
      b.q[1]  = *(const uint4*)(br + hb * 8 + 16);
      acc0 = __builtin_amdgcn_wmma_f32_16x16x32_bf16(false, a0.v, false, b.v, (short)0, acc0, false, false);
      acc1 = __builtin_amdgcn_wmma_f32_16x16x32_bf16(false, a1.v, false, b.v, (short)0, acc1, false, false);
    }
    __syncthreads();
    if (kk + 2 < 72) issue(kk + 2, kk & 1);
  }
  const int col = nb + wn * 16 + lr;
  const float bv = bias[col];
  for (int r = 0; r < 8; ++r) {
    int row0 = mb + wm * 32 + hb * 8 + r;
    if (row0 < HW) {
      float v = acc0[r] + bv; v = v > 0.f ? v : 0.f;
      xact[row0 * 512 + col] = f2bf(v);
    }
    int row1 = mb + wm * 32 + 16 + hb * 8 + r;
    if (row1 < HW) {
      float v = acc1[r] + bv; v = v > 0.f ? v : 0.f;
      xact[row1 * 512 + col] = f2bf(v);
    }
  }
}

// 1x1 heads as one GEMM: M=10000, N=64 (18 score + 36 delta + pad), K=512 -> sd f32
__launch_bounds__(256)
__global__ void head_gemm_kernel(const unsigned short* __restrict__ xact,
                                 const unsigned short* __restrict__ Wc,
                                 const float* __restrict__ bcat,
                                 float* __restrict__ sd) {
  __shared__ __align__(16) unsigned short As[2][64 * 64];
  __shared__ __align__(16) unsigned short Bs[2][64 * 64];
  const int t  = threadIdx.x;
  const int mb = blockIdx.x * 64;
  const int ra = t >> 2;
  const int c8 = (t & 3) * 8;
  int m = mb + ra; if (m >= HW) m = HW - 1;

  const int wave = t >> 5, lane = t & 31;
  const int wm = wave >> 2, wn = wave & 3;
  const int hb = lane >> 4, lr = lane & 15;

  const unsigned lA0 = (unsigned)(size_t)&As[0][0] + (unsigned)((ra * 64 + c8) * 2);
  const unsigned lB0 = (unsigned)(size_t)&Bs[0][0] + (unsigned)((ra * 64 + c8) * 2);

  auto issue = [&](int kk, int buf) {
    const int k0 = kk * 64;
    const unsigned short* ga = xact + m * 512 + k0 + c8;
    const unsigned short* gb = Wc + ra * 512 + k0 + c8;
    const unsigned la = lA0 + (unsigned)buf * 8192u;
    const unsigned lb = lB0 + (unsigned)buf * 8192u;
    async_ld16(la,       ga);
    async_ld16(la + 64u, ga + 32);
    async_ld16(lb,       gb);
    async_ld16(lb + 64u, gb + 32);
  };

  f32x8 acc0 = {}; f32x8 acc1 = {};
  issue(0, 0);
  issue(1, 1);

  for (int kk = 0; kk < 8; ++kk) {
    if (kk < 7) { asm volatile("s_wait_asynccnt 4" ::: "memory"); }
    else        { asm volatile("s_wait_asynccnt 0" ::: "memory"); }
    __syncthreads();
    const unsigned short* Ab = &As[kk & 1][0];
    const unsigned short* Bb = &Bs[kk & 1][0];
#pragma unroll
    for (int ks = 0; ks < 64; ks += 32) {
      Frag a0, a1, b;
      const unsigned short* ar0 = Ab + (wm * 32 + lr) * 64 + ks;
      const unsigned short* ar1 = Ab + (wm * 32 + 16 + lr) * 64 + ks;
      const unsigned short* br  = Bb + (wn * 16 + lr) * 64 + ks;
      a0.q[0] = *(const uint4*)(ar0 + hb * 8);
      a0.q[1] = *(const uint4*)(ar0 + hb * 8 + 16);
      a1.q[0] = *(const uint4*)(ar1 + hb * 8);
      a1.q[1] = *(const uint4*)(ar1 + hb * 8 + 16);
      b.q[0]  = *(const uint4*)(br + hb * 8);
      b.q[1]  = *(const uint4*)(br + hb * 8 + 16);
      acc0 = __builtin_amdgcn_wmma_f32_16x16x32_bf16(false, a0.v, false, b.v, (short)0, acc0, false, false);
      acc1 = __builtin_amdgcn_wmma_f32_16x16x32_bf16(false, a1.v, false, b.v, (short)0, acc1, false, false);
    }
    __syncthreads();
    if (kk + 2 < 8) issue(kk + 2, kk & 1);
  }
  const int col = wn * 16 + lr;
  const float bv = bcat[col];
  for (int r = 0; r < 8; ++r) {
    int row0 = mb + wm * 32 + hb * 8 + r;
    if (row0 < HW) sd[row0 * 64 + col] = acc0[r] + bv;
    int row1 = mb + wm * 32 + 16 + hb * 8 + r;
    if (row1 < HW) sd[row1 * 64 + col] = acc1[r] + bv;
  }
}

// ---------------- proposal path ----------------

__global__ void anchor_post_kernel(const float* __restrict__ sd,
                                   float* __restrict__ fg, float* __restrict__ prop) {
  int n = blockIdx.x * 256 + threadIdx.x;
  if (n >= NANCH) return;
  int m = n / 9, a = n % 9;
  const float* row = sd + m * 64;
  float l0 = row[2 * a], l1 = row[2 * a + 1];
  fg[n] = 1.f / (1.f + expf(l0 - l1));
  float d0 = row[18 + 4 * a + 0], d1 = row[18 + 4 * a + 1];
  float d2 = row[18 + 4 * a + 2], d3 = row[18 + 4 * a + 3];
  float s = c_sizes[a];
  float cx = (float)(m / 100) * 16.f + 8.f;
  float cy = (float)(m % 100) * 16.f + 8.f;
  float dcx = cx + s * d0, dcy = cy + s * d1;
  float dh = s * expf(d2), dw = s * expf(d3);
  prop[n * 4 + 0] = fminf(fmaxf(dcx - dh * 0.5f, 0.f), IMGSZ);
  prop[n * 4 + 1] = fminf(fmaxf(dcy - dw * 0.5f, 0.f), IMGSZ);
  prop[n * 4 + 2] = fminf(fmaxf(dcx + dh * 0.5f, 0.f), IMGSZ);
  prop[n * 4 + 3] = fminf(fmaxf(dcy + dw * 0.5f, 0.f), IMGSZ);
}

__global__ void init_keys_kernel(const float* __restrict__ fg, unsigned long long* __restrict__ keys) {
  int i = blockIdx.x * 256 + threadIdx.x;
  if (i >= NPAD) return;
  unsigned long long k = 0ull;
  if (i < NANCH)
    k = ((unsigned long long)fmono(fg[i]) << 32) | (unsigned)(~(unsigned)i);
  keys[i] = k;
}

__global__ void bitonic_step_kernel(unsigned long long* __restrict__ keys, int j, int kk) {
  unsigned i = blockIdx.x * 256 + threadIdx.x;
  unsigned ixj = i ^ (unsigned)j;
  if (ixj > i) {
    unsigned long long a = keys[i], b = keys[ixj];
    bool descending = ((i & (unsigned)kk) == 0);
    bool sw = descending ? (a < b) : (a > b);
    if (sw) { keys[i] = b; keys[ixj] = a; }
  }
}

__global__ void gather_top_kernel(const unsigned long long* __restrict__ keys,
                                  const float* __restrict__ fg, const float* __restrict__ prop,
                                  float* __restrict__ out) {
  int i = blockIdx.x * 256 + threadIdx.x;
  if (i >= PRENMS) return;
  unsigned long long k = keys[i];
  unsigned idx = ~(unsigned)(k & 0xFFFFFFFFull);
  out[30000 + i] = fg[idx];                 // top_s
  out[i * 4 + 0] = prop[idx * 4 + 0];       // top_b
  out[i * 4 + 1] = prop[idx * 4 + 1];
  out[i * 4 + 2] = prop[idx * 4 + 2];
  out[i * 4 + 3] = prop[idx * 4 + 3];
}

__global__ void nms_mask_kernel(const float* __restrict__ topb, unsigned int* __restrict__ mask) {
  int t = blockIdx.x * 256 + threadIdx.x;
  if (t >= PRENMS * NWRD) return;
  int i = t / NWRD, w = t % NWRD;
  float x1 = topb[i * 4 + 0], y1 = topb[i * 4 + 1];
  float x2 = topb[i * 4 + 2], y2 = topb[i * 4 + 3];
  float ai = (x2 - x1 + 1.f) * (y2 - y1 + 1.f);
  unsigned bits = 0u;
  for (int b = 0; b < 32; ++b) {
    int j = w * 32 + b;
    if (j < PRENMS && j > i) {
      float u1 = topb[j * 4 + 0], v1 = topb[j * 4 + 1];
      float u2 = topb[j * 4 + 2], v2 = topb[j * 4 + 3];
      float aj = (u2 - u1 + 1.f) * (v2 - v1 + 1.f);
      float xx1 = fmaxf(x1, u1), yy1 = fmaxf(y1, v1);
      float xx2 = fminf(x2, u2), yy2 = fminf(y2, v2);
      float wi = fmaxf(0.f, xx2 - xx1 + 1.f);
      float hi = fmaxf(0.f, yy2 - yy1 + 1.f);
      float inter = wi * hi;
      float iou = inter / (ai + aj - inter);
      if (iou > 0.6f) bits |= (1u << b);
    }
  }
  mask[t] = bits;
}

__global__ void nms_scan_kernel(const unsigned int* __restrict__ mask, float* __restrict__ out) {
  __shared__ unsigned int sup[NWRD];
  int t = threadIdx.x;
  if (t < NWRD) sup[t] = 0u;
  __syncthreads();
  for (int i = 0; i < PRENMS; ++i) {
    bool keep_i = ((sup[i >> 5] >> (i & 31)) & 1u) == 0u;
    __syncthreads();
    if (keep_i && t < NWRD) sup[t] |= mask[i * NWRD + t];
    __syncthreads();
  }
  for (int j = t; j < PRENMS; j += 256)
    out[24000 + j] = ((sup[j >> 5] >> (j & 31)) & 1u) ? 0.f : 1.f;
}

// ---------------- loss path ----------------

__global__ void anchor_iou_kernel(const float* __restrict__ gt,
                                  float* __restrict__ maxiou, int* __restrict__ gtfor,
                                  int* __restrict__ label, unsigned long long* __restrict__ gkey) {
  __shared__ float sgt[256];
  __shared__ unsigned long long skey[64];
  int t = threadIdx.x;
  int n = blockIdx.x * 256 + t;
  sgt[t] = gt[t];
  if (t < 64) skey[t] = 0ull;
  __syncthreads();
  if (n < NANCH) {
    float ax1, ay1, ax2, ay2;
    anchor_box(n, ax1, ay1, ax2, ay2);
    bool inside = (ax1 >= 0.f) && (ay1 >= 0.f) && (ax2 <= IMGSZ) && (ay2 <= IMGSZ);
    float aarea = (ax2 - ax1) * (ay2 - ay1);
    float best = -1.f; int bestg = 0;
    for (int g = 0; g < 64; ++g) {
      float iouv = -1.f;
      if (inside) {
        float g0 = sgt[g * 4 + 0], g1 = sgt[g * 4 + 1];
        float g2 = sgt[g * 4 + 2], g3 = sgt[g * 4 + 3];
        float tlx = fmaxf(ax1, g0), tly = fmaxf(ay1, g1);
        float brx = fminf(ax2, g2), bry = fminf(ay2, g3);
        float inter = ((tlx < brx) && (tly < bry)) ? (brx - tlx) * (bry - tly) : 0.f;
        float garea = (g2 - g0) * (g3 - g1);
        iouv = inter / (aarea + garea - inter);
      }
      if (iouv > best) { best = iouv; bestg = g; }
      unsigned long long key = ((unsigned long long)fmono(iouv) << 32) | (unsigned)(~(unsigned)n);
      atomicMax(&skey[g], key);
    }
    maxiou[n] = best;
    gtfor[n]  = bestg;
    int lbl = -1;
    if (inside) {
      if (best > 0.6f) lbl = 1;
      else if (best < 0.2f) lbl = 0;
    }
    label[n] = lbl;
  }
  __syncthreads();
  if (t < 64) atomicMax(&gkey[t], skey[t]);
}

__global__ void gt_assign_kernel(const unsigned long long* __restrict__ gkey, int* __restrict__ label) {
  int g = threadIdx.x;
  if (g >= 64) return;
  unsigned long long k = gkey[g];
  if (k == 0ull) return;
  unsigned n = ~(unsigned)(k & 0xFFFFFFFFull);
  if (n >= NANCH) return;
  float ax1, ay1, ax2, ay2;
  anchor_box((int)n, ax1, ay1, ax2, ay2);
  bool inside = (ax1 >= 0.f) && (ay1 >= 0.f) && (ax2 <= IMGSZ) && (ay2 <= IMGSZ);
  if (inside) label[n] = 1;
}

__global__ void loss_reduce_kernel(const float* __restrict__ sd, const float* __restrict__ gt,
                                   const int* __restrict__ label, const int* __restrict__ gtfor,
                                   float* __restrict__ accum) {
  __shared__ float sgt[256];
  __shared__ float r0[256], r1[256], r2[256];
  int t = threadIdx.x;
  int n = blockIdx.x * 256 + t;
  sgt[t] = gt[t];
  __syncthreads();
  float sl1 = 0.f, ce = 0.f, cnt = 0.f;
  if (n < NANCH) {
    int lbl = label[n];
    int m = n / 9, a = n % 9;
    const float* row = sd + m * 64;
    if (lbl >= 0) {
      cnt = 1.f;
      float l0 = row[2 * a], l1 = row[2 * a + 1];
      float mx = fmaxf(l0, l1);
      float lse = mx + logf(expf(l0 - mx) + expf(l1 - mx));
      ce = lse - ((lbl == 1) ? l1 : l0);
    }
    if (lbl > 0) {
      float ax1, ay1, ax2, ay2;
      anchor_box(n, ax1, ay1, ax2, ay2);
      int g = gtfor[n];
      float g0 = sgt[g * 4 + 0], g1 = sgt[g * 4 + 1];
      float g2 = sgt[g * 4 + 2], g3 = sgt[g * 4 + 3];
      float sh_ = ax2 - ax1 + 1.f, sw_ = ay2 - ay1 + 1.f;
      float scx = ax1 + 0.5f * sh_, scy = ay1 + 0.5f * sw_;
      float dh_ = g2 - g0 + 1.f, dw_ = g3 - g1 + 1.f;
      float dcx = g0 + 0.5f * dh_, dcy = g1 + 0.5f * dw_;
      float tgt[4];
      tgt[0] = (dcx - scx) / sh_;
      tgt[1] = (dcy - scy) / sw_;
      tgt[2] = logf(dh_ / sh_);
      tgt[3] = logf(dw_ / sw_);
      for (int j = 0; j < 4; ++j) {
        float d = row[18 + 4 * a + j] - tgt[j];
        float ad = fabsf(d);
        sl1 += (ad < 1.f / 9.f) ? (4.5f * ad * ad) : (ad - 1.f / 18.f);
      }
    }
  }
  r0[t] = sl1; r1[t] = ce; r2[t] = cnt;
  __syncthreads();
  for (int s = 128; s > 0; s >>= 1) {
    if (t < s) { r0[t] += r0[t + s]; r1[t] += r1[t + s]; r2[t] += r2[t + s]; }
    __syncthreads();
  }
  if (t == 0) {
    atomicAdd(&accum[0], r0[0]);
    atomicAdd(&accum[1], r1[0]);
    atomicAdd(&accum[2], r2[0]);
  }
}

__global__ void finalize_kernel(const float* __restrict__ accum, float* __restrict__ out) {
  float nv = fmaxf(accum[2], 1.f);
  out[36000] = (accum[0] + accum[1]) / nv;
}

// ---------------- launcher ----------------

extern "C" void kernel_launch(void* const* d_in, const int* in_sizes, int n_in,
                              void* d_out, int out_size, void* d_ws, size_t ws_size,
                              hipStream_t stream) {
  (void)in_sizes; (void)n_in; (void)out_size; (void)ws_size;
  const float* feat    = (const float*)d_in[0];
  const float* conv_w  = (const float*)d_in[1];
  const float* conv_b  = (const float*)d_in[2];
  const float* score_w = (const float*)d_in[3];
  const float* score_b = (const float*)d_in[4];
  const float* delta_w = (const float*)d_in[5];
  const float* delta_b = (const float*)d_in[6];
  const float* gt      = (const float*)d_in[7];
  float* out = (float*)d_out;

  char* base = (char*)d_ws;
  size_t off = 0;
  auto alloc = [&](size_t bytes) -> void* {
    void* r = base + off;
    off += bytes;
    off = (off + 255) & ~(size_t)255;
    return r;
  };

  unsigned short* fp   = (unsigned short*)alloc((size_t)102 * 102 * 512 * 2);
  unsigned short* Bt   = (unsigned short*)alloc((size_t)512 * KDIM * 2);
  unsigned short* Wc   = (unsigned short*)alloc((size_t)64 * 512 * 2);
  float*  bcat         = (float*)alloc(64 * 4);
  unsigned short* xact = (unsigned short*)alloc((size_t)HW * 512 * 2);
  float*  sd           = (float*)alloc((size_t)HW * 64 * 4);
  float*  fg           = (float*)alloc((size_t)NANCH * 4);
  float*  prop         = (float*)alloc((size_t)NANCH * 16);
  unsigned long long* keys = (unsigned long long*)alloc((size_t)NPAD * 8);
  unsigned int* mask   = (unsigned int*)alloc((size_t)PRENMS * NWRD * 4);
  float*  maxiou       = (float*)alloc((size_t)NANCH * 4);
  int*    gtfor        = (int*)alloc((size_t)NANCH * 4);
  int*    label        = (int*)alloc((size_t)NANCH * 4);
  unsigned long long* gkey = (unsigned long long*)alloc(64 * 8);
  float*  accum        = (float*)alloc(16);

  zero_misc_kernel<<<1, 64, 0, stream>>>(gkey, accum);
  pad_feat_kernel<<<(102 * 102 * 512 + 255) / 256, 256, 0, stream>>>(feat, fp);
  prep_bt_kernel<<<(512 * KDIM + 255) / 256, 256, 0, stream>>>(conv_w, Bt);
  prep_wcat_kernel<<<(64 * 512 + 255) / 256, 256, 0, stream>>>(score_w, score_b, delta_w, delta_b, Wc, bcat);

  conv_gemm_kernel<<<dim3(157, 8), 256, 0, stream>>>(fp, Bt, conv_b, xact);
  head_gemm_kernel<<<157, 256, 0, stream>>>(xact, Wc, bcat, sd);

  anchor_post_kernel<<<(NANCH + 255) / 256, 256, 0, stream>>>(sd, fg, prop);
  init_keys_kernel<<<NPAD / 256, 256, 0, stream>>>(fg, keys);
  for (int kk = 2; kk <= NPAD; kk <<= 1)
    for (int j = kk >> 1; j >= 1; j >>= 1)
      bitonic_step_kernel<<<NPAD / 256, 256, 0, stream>>>(keys, j, kk);

  gather_top_kernel<<<(PRENMS + 255) / 256, 256, 0, stream>>>(keys, fg, prop, out);
  nms_mask_kernel<<<(PRENMS * NWRD + 255) / 256, 256, 0, stream>>>(out, mask);
  nms_scan_kernel<<<1, 256, 0, stream>>>(mask, out);

  anchor_iou_kernel<<<(NANCH + 255) / 256, 256, 0, stream>>>(gt, maxiou, gtfor, label, gkey);
  gt_assign_kernel<<<1, 64, 0, stream>>>(gkey, label);
  loss_reduce_kernel<<<(NANCH + 255) / 256, 256, 0, stream>>>(sd, gt, label, gtfor, accum);
  finalize_kernel<<<1, 1, 0, stream>>>(accum, out);
}
